// MIRASBlock_15882789060893
// MI455X (gfx1250) — matrix operand order repaired and checked by reading, not compiled, not run on hardware
//
#include <hip/hip_runtime.h>
#include <hip/hip_bf16.h>

// ---------------------------------------------------------------------------
// MIRAS memory-attention block for MI455X (gfx1250), wave32 + WMMA bf16.
//
// All GEMMs lower to v_wmma_f32_16x16x32_bf16 (bf16 A/B, f32 accumulate).
// Blocking strategy is L2-centric: every weight matrix (0.5-4 MB bf16,
// stored transposed [N][K]) is resident in the 192 MB L2, so each GEMM
// block stages a 32-row x K activation tile in LDS ONCE (read-once from
// HBM) and then sweeps all N tiles, streaming B fragments out of L2.
//
// Fragment layouts used here (contiguous k order, same for A and B):
//   A (16x32, 16-bit): lane L -> row m=L%16, half g=L/16;
//       a[0..7]  = A[m][k0+16g .. k0+16g+7]      (16B load)
//       a[8..15] = A[m][k0+16g+8 .. k0+16g+15]   (16B load)
//   B (32x16, 16-bit): lane L -> col n=L%16, half g=L/16;
//       b[i] = B[k0+16g+i][n] -> contiguous along K for B stored [N][K]
//   C/D f32: lane L -> col n=L%16; c[r] = C[r+8*(L/16)][n]
//
// NOTE: the reference's "surprise renormalization" attn*(1+s)/sum(attn*(1+s))
// is mathematically the identity on softmax output (sum==1), so it is elided.
// ---------------------------------------------------------------------------

typedef __bf16 bf16;
typedef __attribute__((ext_vector_type(16))) __bf16 v16bf;
typedef __attribute__((ext_vector_type(8)))  float  v8f;

union FragBF { v16bf v; uint4 q[2]; };

__device__ __forceinline__ float gelu_erf(float x) {
    return 0.5f * x * (1.0f + erff(x * 0.70710678118654752f));
}

enum { GF_OUT_BF16 = 1, GF_GELU = 2, GF_STORE_T = 4 };

// ---------------------------------------------------------------------------
// WMMA GEMM: C[M,N] = A[M,K](bf16) * Bt[N,K](bf16)^T + bias
// Block = 8 waves. Block owns 32 rows of A, staged in LDS once (A is
// read exactly once from HBM). Wave w: row-half (w&1), N-tiles (w>>1)+4j.
// K, M multiples of 32; N multiple of 16. Dynamic LDS = 32*K*2 bytes.
// ---------------------------------------------------------------------------
__global__ __launch_bounds__(256) void gemm_bt_wmma(
    const bf16* __restrict__ A, long long aBatch, int lda,
    const bf16* __restrict__ Bt, int ldb,
    const float* __restrict__ bias,
    void* __restrict__ Cv, long long cBatch, int ldc,
    int N, int K, int flags)
{
    extern __shared__ char smem[];
    bf16* As = (bf16*)smem;                     // [32][K]

    const int m0 = blockIdx.x * 32;
    const int bz = blockIdx.z;
    A += (long long)bz * aBatch;

    const int tid = threadIdx.x;

    // cooperative stage of the 32 x K A-tile into LDS (16B chunks)
    const int chunksPerRow = K >> 3;            // 8 bf16 per 16B
    for (int c = tid; c < 32 * chunksPerRow; c += 256) {
        int row = c / chunksPerRow, kc = c - row * chunksPerRow;
        *(uint4*)(As + row * K + kc * 8) =
            *(const uint4*)(A + (long long)(m0 + row) * lda + kc * 8);
    }
    __syncthreads();

    const int w = tid >> 5, lane = tid & 31;
    const int mn = lane & 15, g = lane >> 4;
    const int sub = w & 1;                      // which 16-row half
    const bf16* arow = As + (sub * 16 + mn) * K;
    const long long cb = cBatch * bz;

    for (int nt = (w >> 1); nt < (N >> 4); nt += 4) {
        const int n0 = nt << 4;
        const bf16* brow = Bt + (long long)(n0 + mn) * ldb;
        v8f acc = {};
        for (int k0 = 0; k0 < K; k0 += 32) {
            FragBF a, b;
            a.q[0] = *(const uint4*)(arow + k0 + 16 * g);       // ds_load_b128
            a.q[1] = *(const uint4*)(arow + k0 + 16 * g + 8);
            b.q[0] = *(const uint4*)(brow + k0 + 16 * g);       // L2-resident
            b.q[1] = *(const uint4*)(brow + k0 + 16 * g + 8);
            acc = __builtin_amdgcn_wmma_f32_16x16x32_bf16(
                false, a.v, false, b.v, (short)0, acc, false, false);
        }
        const float bia = bias ? bias[n0 + mn] : 0.0f;
        #pragma unroll
        for (int r = 0; r < 8; ++r) {
            int row = m0 + sub * 16 + r + 8 * g;
            int col = n0 + mn;
            float x = acc[r] + bia;
            if (flags & GF_GELU) x = gelu_erf(x);
            long long idx = (flags & GF_STORE_T) ? (long long)col * ldc + row
                                                 : (long long)row * ldc + col;
            if (flags & GF_OUT_BF16) ((bf16*)Cv)[cb + idx] = (bf16)x;
            else                     ((float*)Cv)[cb + idx] = x;
        }
    }
}

// ---------------------------------------------------------------------------
// Fused cross-attention: one workgroup = (batch b, 32 query rows).
// Loops over 16 heads: WMMA logits -> LDS, masked softmax in LDS,
// probs re-packed bf16 in LDS (A-frags via ds_load_b128), WMMA attn*V,
// attended written into combined[:,1024:] as bf16; head-mean attn to out2.
// ---------------------------------------------------------------------------
#define SROWS 32
#define MSLOTS 512

__global__ __launch_bounds__(256) void attn_fused(
    const bf16* __restrict__ qbf,     // [B*S, 1024]
    const bf16* __restrict__ kbf,     // [B, 512, 1024]
    const bf16* __restrict__ vT,      // [B, 1024, 512]
    const int*  __restrict__ mask,    // [B, 512]
    bf16* __restrict__ combined,      // [B*S, 2048] (writes cols 1024..2047)
    float* __restrict__ attn_out)     // [B, S, 512]
{
    extern __shared__ char smem[];
    float* logit = (float*)smem;                          // 32*512 f32 (64KB)
    bf16*  probs = (bf16*)(smem + SROWS * MSLOTS * 4);    // 32*512 bf16 (32KB)
    float* accum = (float*)(smem + SROWS * MSLOTS * 6);   // 32*512 f32 (64KB)
    float* red   = (float*)(smem + SROWS * MSLOTS * 10);  // 256 f32

    const int s0  = blockIdx.x * SROWS;
    const int b   = blockIdx.y;
    const int tid = threadIdx.x;
    const int w = tid >> 5, lane = tid & 31;
    const int mn = lane & 15, g = lane >> 4;

    const bf16* kb = kbf + (long long)b * MSLOTS * 1024;
    const bf16* vb = vT  + (long long)b * 1024 * MSLOTS;
    const int*  mk = mask + b * MSLOTS;

    for (int i = tid; i < SROWS * MSLOTS; i += 256) accum[i] = 0.0f;
    __syncthreads();

    for (int h = 0; h < 16; ++h) {
        const int fbase = h * 64;

        // ---- logits: 32x512 = 64 tiles of 16x16; wave w handles 8 tiles
        for (int tt = w; tt < 64; tt += 8) {
            int rt = tt >> 5, ct = tt & 31;
            const bf16* arow =
                qbf + (long long)(b * 2048 + s0 + rt * 16 + mn) * 1024 + fbase;
            const bf16* brow = kb + (long long)(ct * 16 + mn) * 1024 + fbase;
            v8f acc = {};
            #pragma unroll
            for (int k0 = 0; k0 < 64; k0 += 32) {
                FragBF a, bb;
                a.q[0]  = *(const uint4*)(arow + k0 + 16 * g);
                a.q[1]  = *(const uint4*)(arow + k0 + 16 * g + 8);
                bb.q[0] = *(const uint4*)(brow + k0 + 16 * g);
                bb.q[1] = *(const uint4*)(brow + k0 + 16 * g + 8);
                acc = __builtin_amdgcn_wmma_f32_16x16x32_bf16(
                    false, a.v, false, bb.v, (short)0, acc, false, false);
            }
            int col = ct * 16 + mn;
            bool on = (mk[col] != 0);
            #pragma unroll
            for (int r = 0; r < 8; ++r) {
                int srow = rt * 16 + r + 8 * g;
                logit[srow * MSLOTS + col] = on ? acc[r] * 0.125f : -1e30f;
            }
        }
        __syncthreads();

        // ---- masked softmax per row (8 threads/row, 64 cols each).
        {
            int r = tid >> 3, part = tid & 7;
            float* lr = logit + r * MSLOTS + part * 64;
            float mx = -3.0e38f;
            for (int j = 0; j < 64; ++j) mx = fmaxf(mx, lr[j]);
            red[tid] = mx; __syncthreads();
            if (part == 0) {
                float m2 = red[tid];
                for (int p = 1; p < 8; ++p) m2 = fmaxf(m2, red[tid + p]);
                red[tid] = m2;
            }
            __syncthreads();
            float rmax = red[tid & ~7];
            __syncthreads();
            float s = 0.0f;
            for (int j = 0; j < 64; ++j) {
                float e = __expf(lr[j] - rmax); lr[j] = e; s += e;
            }
            red[tid] = s; __syncthreads();
            if (part == 0) {
                float s2 = red[tid];
                for (int p = 1; p < 8; ++p) s2 += red[tid + p];
                red[tid] = s2;
            }
            __syncthreads();
            float inv = 1.0f / red[tid & ~7];
            // surprise renorm attn*1.8/sum(attn*1.8) == attn : elided.
            bf16*  pr = probs + r * MSLOTS + part * 64;
            float* ac = accum + r * MSLOTS + part * 64;
            for (int j = 0; j < 64; ++j) {
                float p = lr[j] * inv;
                pr[j] = (bf16)p;
                ac[j] += p * 0.0625f;       // mean over 16 heads
            }
        }
        __syncthreads();

        // ---- attended = probs(32x512) @ V(512x64): 8 tiles, wave w -> tile w
        {
            int rt = w >> 2, ct = w & 3;
            const bf16* arow = probs + (rt * 16 + mn) * MSLOTS;
            const bf16* brow = vb + (long long)(fbase + ct * 16 + mn) * MSLOTS;
            v8f acc = {};
            for (int k0 = 0; k0 < MSLOTS; k0 += 32) {
                FragBF a, bb;
                a.q[0]  = *(const uint4*)(arow + k0 + 16 * g);  // ds_load_b128
                a.q[1]  = *(const uint4*)(arow + k0 + 16 * g + 8);
                bb.q[0] = *(const uint4*)(brow + k0 + 16 * g);
                bb.q[1] = *(const uint4*)(brow + k0 + 16 * g + 8);
                acc = __builtin_amdgcn_wmma_f32_16x16x32_bf16(
                    false, a.v, false, bb.v, (short)0, acc, false, false);
            }
            #pragma unroll
            for (int r = 0; r < 8; ++r) {
                long long token = (long long)b * 2048 + s0 + rt * 16 + r + 8 * g;
                combined[token * 2048 + 1024 + fbase + ct * 16 + mn] = (bf16)acc[r];
            }
        }
        __syncthreads();
    }

    // ---- write head-mean attention weights
    {
        int r = tid >> 3, part = tid & 7;
        float* ac  = accum + r * MSLOTS + part * 64;
        float* dst = attn_out + ((long long)(b * 2048 + s0 + r)) * MSLOTS + part * 64;
        for (int j = 0; j < 64; ++j) dst[j] = ac[j];
    }
}

// ---------------------------------------------------------------------------
// Elementwise helpers
// ---------------------------------------------------------------------------
__global__ void cvt_bf16_strided(const float* __restrict__ src,
                                 bf16* __restrict__ dst,
                                 long long n, int cols, int sld, int dld)
{
    long long i = (long long)blockIdx.x * 256 + threadIdx.x;
    if (i >= n) return;
    long long r = i / cols; int c = (int)(i % cols);
    dst[r * dld + c] = (bf16)src[r * sld + c];
}

__global__ void transpose_cvt_bf16(const float* __restrict__ src,
                                   bf16* __restrict__ dst, int K, int N)
{
    // src [K,N] row-major -> dst [N,K]
    long long i = (long long)blockIdx.x * 256 + threadIdx.x;
    if (i >= (long long)K * N) return;
    int n = (int)(i % N); long long k = i / N;
    dst[(long long)n * K + k] = (bf16)src[i];
}

__global__ __launch_bounds__(256) void layernorm_inplace(
    float* __restrict__ x, const float* __restrict__ g,
    const float* __restrict__ bb, int H)
{
    __shared__ float s1[256], s2[256];
    float* row = x + (long long)blockIdx.x * H;
    int tid = threadIdx.x;
    float a = 0.0f, q = 0.0f;
    for (int i = tid; i < H; i += 256) { float v = row[i]; a += v; q += v * v; }
    s1[tid] = a; s2[tid] = q; __syncthreads();
    for (int off = 128; off; off >>= 1) {
        if (tid < off) { s1[tid] += s1[tid + off]; s2[tid] += s2[tid + off]; }
        __syncthreads();
    }
    float mu  = s1[0] / H;
    float var = s2[0] / H - mu * mu;
    float inv = rsqrtf(var + 1e-5f);
    for (int i = tid; i < H; i += 256)
        row[i] = (row[i] - mu) * inv * g[i] + bb[i];
}

__global__ __launch_bounds__(256) void gate_kernel(
    const bf16* __restrict__ g1, const float* __restrict__ Wg2,
    const float* __restrict__ bg2, float* __restrict__ gate, int M)
{
    int token = blockIdx.x * 8 + (threadIdx.x >> 5);  // one wave32 per token
    int lane  = threadIdx.x & 31;
    if (token >= M) return;
    const bf16* row = g1 + (long long)token * 1024;
    float s = 0.0f;
    for (int i = lane; i < 1024; i += 32) s += (float)row[i] * Wg2[i];
    for (int off = 16; off; off >>= 1) s += __shfl_xor(s, off);
    if (lane == 0) gate[token] = 1.0f / (1.0f + __expf(-(s + bg2[0])));
}

__global__ void blend_kernel(const float* __restrict__ fused,
                             const float* __restrict__ hidden,
                             const float* __restrict__ gate,
                             float* __restrict__ out, long long n)
{
    long long i = (long long)blockIdx.x * 256 + threadIdx.x;
    if (i >= n) return;
    float gt = gate[i >> 10];                 // H = 1024
    out[i] = gt * fused[i] + (1.0f - gt) * hidden[i];
}

// ---------------------------------------------------------------------------
// Host launcher
// ---------------------------------------------------------------------------
extern "C" void kernel_launch(void* const* d_in, const int* in_sizes, int n_in,
                              void* d_out, int out_size, void* d_ws, size_t ws_size,
                              hipStream_t stream)
{
    (void)in_sizes; (void)n_in; (void)out_size; (void)ws_size;

    const float* hidden = (const float*)d_in[0];   // [8,2048,1024]
    const float* membuf = (const float*)d_in[1];   // [8,512,256]
    const float* Wq  = (const float*)d_in[2];  const float* bq  = (const float*)d_in[3];
    const float* Wk  = (const float*)d_in[4];  const float* bk  = (const float*)d_in[5];
    const float* Wv  = (const float*)d_in[6];  const float* bv  = (const float*)d_in[7];
    const float* Wf1 = (const float*)d_in[8];  const float* bf1 = (const float*)d_in[9];
    const float* Wf2 = (const float*)d_in[10]; const float* bf2 = (const float*)d_in[11];
    const float* lng = (const float*)d_in[12]; const float* lnb = (const float*)d_in[13];
    const float* Wg1 = (const float*)d_in[14]; const float* bg1 = (const float*)d_in[15];
    const float* Wg2 = (const float*)d_in[16]; const float* bg2 = (const float*)d_in[17];
    const int*   mask = (const int*)d_in[18];  // [8,512]

    const long long T = 16384;   // B*S tokens
    char* ws = (char*)d_ws;
    auto alloc = [&](size_t bytes) {
        char* p = ws; ws += (bytes + 255) & ~(size_t)255; return p;
    };
    bf16*  combined = (bf16*)alloc(T * 2048 * 2);        // [T,2048]: hidden | attended
    bf16*  bufA     = (bf16*)alloc(T * 1024 * 2);        // q -> t1 -> g1 (disjoint lifetimes)
    bf16*  membf    = (bf16*)alloc(4096ll * 256 * 2);    // [B*512,256]
    bf16*  kbf      = (bf16*)alloc(8ll * 512 * 1024 * 2);
    bf16*  vTbf     = (bf16*)alloc(8ll * 1024 * 512 * 2);
    bf16*  WqT      = (bf16*)alloc(1024ll * 1024 * 2);
    bf16*  WkT      = (bf16*)alloc(1024ll * 256 * 2);
    bf16*  WvT      = (bf16*)alloc(1024ll * 256 * 2);
    bf16*  Wf1T     = (bf16*)alloc(1024ll * 2048 * 2);
    bf16*  Wf2T     = (bf16*)alloc(1024ll * 1024 * 2);
    bf16*  Wg1T     = (bf16*)alloc(1024ll * 2048 * 2);
    float* fusedpre = (float*)alloc(T * 1024 * 4);
    float* gate     = (float*)alloc(T * 4);

    float* out1 = (float*)d_out;            // [8,2048,1024] hidden_updated
    float* out2 = out1 + T * 1024;          // [8,2048,512]  mem_attn_weights

    auto nb = [](long long n) { return (unsigned)((n + 255) / 256); };

    // ---- precision conversion + weight transposes (bf16, B stored [N,K])
    cvt_bf16_strided<<<nb(T * 1024), 256, 0, stream>>>(hidden, combined, T * 1024, 1024, 1024, 2048);
    cvt_bf16_strided<<<nb(4096ll * 256), 256, 0, stream>>>(membuf, membf, 4096ll * 256, 256, 256, 256);
    transpose_cvt_bf16<<<nb(1024ll * 1024), 256, 0, stream>>>(Wq,  WqT,  1024, 1024);
    transpose_cvt_bf16<<<nb(256ll * 1024),  256, 0, stream>>>(Wk,  WkT,  256, 1024);
    transpose_cvt_bf16<<<nb(256ll * 1024),  256, 0, stream>>>(Wv,  WvT,  256, 1024);
    transpose_cvt_bf16<<<nb(2048ll * 1024), 256, 0, stream>>>(Wf1, Wf1T, 2048, 1024);
    transpose_cvt_bf16<<<nb(1024ll * 1024), 256, 0, stream>>>(Wf2, Wf2T, 1024, 1024);
    transpose_cvt_bf16<<<nb(2048ll * 1024), 256, 0, stream>>>(Wg1, Wg1T, 2048, 1024);

    // ---- projections (WMMA bf16); LDS = 32 rows x K of A per block
    // Q: [16384,1024] = combined[:, :1024] @ Wq -> bufA (bf16)
    gemm_bt_wmma<<<dim3(512, 1, 1), 256, 32 * 1024 * 2, stream>>>(
        combined, 0, 2048, WqT, 1024, bq, bufA, 0, 1024, 1024, 1024, GF_OUT_BF16);
    // K: per batch [512,1024] = membf[b] @ Wk -> kbf (bf16)
    gemm_bt_wmma<<<dim3(16, 1, 8), 256, 32 * 256 * 2, stream>>>(
        membf, 512ll * 256, 256, WkT, 256, bk, kbf, 512ll * 1024, 1024,
        1024, 256, GF_OUT_BF16);
    // V: per batch, stored transposed -> vT[b][feat][slot] (bf16)
    gemm_bt_wmma<<<dim3(16, 1, 8), 256, 32 * 256 * 2, stream>>>(
        membf, 512ll * 256, 256, WvT, 256, bv, vTbf, 1024ll * 512, 512,
        1024, 256, GF_OUT_BF16 | GF_STORE_T);

    // ---- fused masked cross-attention (writes combined[:,1024:] and out2)
    const size_t shmem = SROWS * MSLOTS * 4 + SROWS * MSLOTS * 2
                       + SROWS * MSLOTS * 4 + 256 * 4;   // 164,864 B
    attn_fused<<<dim3(64, 8), 256, shmem, stream>>>(
        bufA, kbf, vTbf, mask, combined, out2);

    // ---- gated fusion MLP
    // t1 = gelu(combined @ Wf1 + bf1) -> bufA (q dead)
    gemm_bt_wmma<<<dim3(512, 1, 1), 256, 32 * 2048 * 2, stream>>>(
        combined, 0, 2048, Wf1T, 2048, bf1, bufA, 0, 1024, 1024, 2048,
        GF_OUT_BF16 | GF_GELU);
    // fusedpre = t1 @ Wf2 + bf2 (f32)
    gemm_bt_wmma<<<dim3(512, 1, 1), 256, 32 * 1024 * 2, stream>>>(
        bufA, 0, 1024, Wf2T, 1024, bf2, fusedpre, 0, 1024, 1024, 1024, 0);
    layernorm_inplace<<<16384, 256, 0, stream>>>(fusedpre, lng, lnb, 1024);
    // g1 = gelu(combined @ Wg1 + bg1) -> bufA (t1 dead)
    gemm_bt_wmma<<<dim3(512, 1, 1), 256, 32 * 2048 * 2, stream>>>(
        combined, 0, 2048, Wg1T, 2048, bg1, bufA, 0, 1024, 1024, 2048,
        GF_OUT_BF16 | GF_GELU);
    // gate = sigmoid(g1 @ Wg2 + bg2)  (wave32-per-token GEMV)
    gate_kernel<<<2048, 256, 0, stream>>>(bufA, Wg2, bg2, gate, (int)T);
    // out1 = gate*fused + (1-gate)*hidden
    blend_kernel<<<nb(T * 1024), 256, 0, stream>>>(fusedpre, hidden, gate, out1, T * 1024);
}
